// Transformer_15556371546777
// MI455X (gfx1250) — matrix operand order repaired and checked by reading, not compiled, not run on hardware
//
#include <hip/hip_runtime.h>
#include <hip/hip_bf16.h>

// ---------------------------------------------------------------------------
// CDNA5 / gfx1250 TransformerConv (PyG semantics, heads=1) — wave32 + WMMA.
// GEMMs use v_wmma_f32_16x16x32_bf16 (bf16 A/B from f32, f32 accumulate).
// B operands are pre-swizzled into WMMA fragment-major layout so each lane's
// 16-element fragment is one aligned 32-byte load (2x global_load_b128).
// Edge softmax/aggregation use native f32/u32 global atomics (L2-resident).
// ---------------------------------------------------------------------------

typedef __attribute__((ext_vector_type(16))) __bf16 v16bf;
typedef __attribute__((ext_vector_type(4)))  __bf16 v4bf;
typedef __attribute__((ext_vector_type(8)))  float  v8f;

#define N_NODES 100000
#define N_EDGES 1600000
#define OUTD    64        // per-projection output dim
#define FUSED   256       // Q|K|V|S fused columns

__device__ __forceinline__ __bf16 f2bf(float f) { return (__bf16)f; }

// order-preserving float<->uint encoding for atomicMax-based segment max
__device__ __forceinline__ unsigned encf(float f) {
    unsigned u = __float_as_uint(f);
    return (u & 0x80000000u) ? ~u : (u | 0x80000000u);
}
__device__ __forceinline__ float decf(unsigned u) {
    return __uint_as_float((u & 0x80000000u) ? (u ^ 0x80000000u) : ~u);
}

// ---------------------------------------------------------------------------
// Utility fills (workspace is poisoned by the harness; re-init every call)
// ---------------------------------------------------------------------------
__global__ void fill_f32(float* p, float v, int n) {
    int i = blockIdx.x * blockDim.x + threadIdx.x;
    if (i < n) p[i] = v;
}
__global__ void fill_u32(unsigned* p, unsigned v, int n) {
    int i = blockIdx.x * blockDim.x + threadIdx.x;
    if (i < n) p[i] = v;
}

// ---------------------------------------------------------------------------
// Pack Wq|Wk|Wv|Ws (each [d_in,64] f32 row-major) into WMMA-fragment-major
// bf16 layout:
//   Wswz[((ntile*KSTEPS + ks)*32 + lane)*16 + j]
//     = W[K = ks*32 + (lane>>4)*16 + j][col = ntile*16 + (lane&15)]
// so each lane's B fragment for one 16x16x32 WMMA is 16 contiguous bf16.
// Also packs biases -> bcat f32 [256].
// ---------------------------------------------------------------------------
__global__ void pack_weights(const float* __restrict__ Wq, const float* __restrict__ Wk,
                             const float* __restrict__ Wv, const float* __restrict__ Ws,
                             const float* __restrict__ bq, const float* __restrict__ bk,
                             const float* __restrict__ bv, const float* __restrict__ bs,
                             __bf16* __restrict__ Wswz, float* __restrict__ bcat, int d_in) {
    int i = blockIdx.x * blockDim.x + threadIdx.x;
    int ksteps = d_in >> 5;                 // DIN/32
    int total  = 16 * ksteps * 32 * 16;     // == d_in * 256
    if (i >= total) return;

    int j     = i & 15;
    int lane  = (i >> 4) & 31;
    int ks    = (i >> 9) % ksteps;
    int ntile = i / (ksteps << 9);

    int col   = ntile * 16 + (lane & 15);
    int K     = ks * 32 + (lane >> 4) * 16 + j;

    int grp = col >> 6;                     // 0:q 1:k 2:v 3:s
    int jj  = col & 63;
    const float* W = (grp == 0) ? Wq : (grp == 1) ? Wk : (grp == 2) ? Wv : Ws;
    Wswz[i] = f2bf(W[K * OUTD + jj]);

    if (i < FUSED) {                        // bias pack (one thread per fused col)
        int g2 = i >> 6, j2 = i & 63;
        const float* b = (g2 == 0) ? bq : (g2 == 1) ? bk : (g2 == 2) ? bv : bs;
        bcat[i] = b[j2];
    }
}

// ---------------------------------------------------------------------------
// Fused QKVS GEMM:  Y[N,256] = X[N,d_in] @ W[d_in,256] + bcat
// Block = 128 threads (4 wave32); each wave owns one 16x16 tile.
// grid = (N/16, 4); wave's ntile = blockIdx.y*4 + waveId (16 ntiles = 256 cols)
// ---------------------------------------------------------------------------
template <int DIN>
__global__ void qkvs_gemm(const float* __restrict__ X,
                          const __bf16* __restrict__ Wswz,
                          const float*  __restrict__ bcat,
                          float* __restrict__ Y) {
    constexpr int KSTEPS = DIN / 32;
    __shared__ __bf16 ldsx[16 * DIN];

    const int mtile = blockIdx.x;
    const int wave  = threadIdx.x >> 5;
    const int lane  = threadIdx.x & 31;

    // cooperative A-tile load: X[mtile*16 .. +16, 0..DIN) -> bf16 LDS
    // float4 global reads, packed bf16x4 (8B) LDS writes
    {
        const float4* X4 = (const float4*)(X + (size_t)mtile * 16 * DIN);
        v4bf* L4 = (v4bf*)ldsx;
#pragma unroll
        for (int i = threadIdx.x; i < 4 * DIN; i += 128) {
            float4 f = X4[i];
            v4bf o;
            o[0] = f2bf(f.x); o[1] = f2bf(f.y); o[2] = f2bf(f.z); o[3] = f2bf(f.w);
            L4[i] = o;
        }
    }
    __syncthreads();

    const int ntile = blockIdx.y * 4 + wave;   // 0..15
    const int gcol0 = ntile * 16;
    const int mrow  = lane & 15;
    const int khalf = lane >> 4;               // 0 or 1

    // per-lane fragment base in swizzled weights (32B-aligned)
    const v16bf* Wfrag =
        (const v16bf*)(Wswz + ((size_t)(ntile * KSTEPS) * 32 + lane) * 16);

    v8f acc = {};
#pragma unroll
    for (int ks = 0; ks < KSTEPS; ++ks) {
        // A fragment (16x32 bf16): lanes 0-15 hold K {0..15},
        // lanes 16-31 hold K {16..31} of row mrow. Two 16B LDS loads.
        v16bf a;
        const __bf16* xr = &ldsx[mrow * DIN + ks * 32];
#pragma unroll
        for (int i = 0; i < 8; ++i) {
            a[i]     = xr[16 * khalf + i];
            a[i + 8] = xr[16 * khalf + 8 + i];
        }
        // B fragment: one aligned 32B load from fragment-major weights
        v16bf b = Wfrag[(size_t)ks * 32];

        acc = __builtin_amdgcn_wmma_f32_16x16x32_bf16(
            /*neg_a=*/false, a, /*neg_b=*/false, b,
            /*c_mod=*/(short)0, acc, /*reuse_a=*/false, /*reuse_b=*/false);
    }

    // C/D layout: VGPR r -> row r (lanes 0-15) / row r+8 (lanes 16-31), col = lane&15
    const int   coln  = lane & 15;
    const float bia   = bcat[gcol0 + coln];
    const int   rbase = (lane < 16) ? 0 : 8;
#pragma unroll
    for (int r = 0; r < 8; ++r) {
        int row = mtile * 16 + rbase + r;
        Y[(size_t)row * FUSED + gcol0 + coln] = acc[r] + bia;
    }
}

// ---------------------------------------------------------------------------
// Per-edge attention score: score[e] = dot(q[dst], k[src]) / sqrt(64)
// Y layout per node: cols [0,64)=q, [64,128)=k, [128,192)=v, [192,256)=s
// ---------------------------------------------------------------------------
__global__ void edge_scores(const int* __restrict__ src, const int* __restrict__ dst,
                            const float* __restrict__ Y, float* __restrict__ score,
                            int nE, float scale) {
    int e = blockIdx.x * blockDim.x + threadIdx.x;
    if (e >= nE) return;
    int s = src[e], d = dst[e];
    const float4* q = (const float4*)(Y + (size_t)d * FUSED);
    const float4* k = (const float4*)(Y + (size_t)s * FUSED + 64);
    float acc = 0.f;
#pragma unroll
    for (int i = 0; i < 16; ++i) {
        float4 a = q[i], b = k[i];
        acc += a.x * b.x + a.y * b.y + a.z * b.z + a.w * b.w;
    }
    score[e] = acc * scale;
}

// segment max over dst (order-preserving encoded atomicMax -> native u32 atomic)
__global__ void seg_max(const int* __restrict__ dst, const float* __restrict__ score,
                        unsigned* __restrict__ menc, int nE) {
    int e = blockIdx.x * blockDim.x + threadIdx.x;
    if (e >= nE) return;
    atomicMax(&menc[dst[e]], encf(score[e]));
}

// e = exp(score - m[dst]); denom[dst] += e
__global__ void exp_sum(const int* __restrict__ dst, const float* __restrict__ score,
                        const unsigned* __restrict__ menc, float* __restrict__ evals,
                        float* __restrict__ denom, int nE) {
    int e = blockIdx.x * blockDim.x + threadIdx.x;
    if (e >= nE) return;
    int d = dst[e];
    float ev = __expf(score[e] - decf(menc[d]));
    evals[e] = ev;
    atomicAdd(&denom[d], ev);
}

// agg[dst] += alpha * v[src]  — one wave32 per edge, 2 channels per lane
__global__ void scatter_agg(const int* __restrict__ src, const int* __restrict__ dst,
                            const float* __restrict__ Y, const float* __restrict__ evals,
                            const float* __restrict__ denom, float* __restrict__ agg, int nE) {
    int gid  = blockIdx.x * blockDim.x + threadIdx.x;
    int e    = gid >> 5;
    int lane = gid & 31;
    if (e >= nE) return;
    int s = src[e], d = dst[e];
    float alpha = evals[e] / fmaxf(denom[d], 1e-16f);
    const float* v = Y + (size_t)s * FUSED + 128;
    float*       o = agg + (size_t)d * OUTD;
    atomicAdd(&o[lane],      alpha * v[lane]);
    atomicAdd(&o[lane + 32], alpha * v[lane + 32]);
}

// out = act(agg + skip);  mode 0 = relu (hidden layers), 1 = sigmoid (final)
__global__ void epilogue(const float* __restrict__ agg, const float* __restrict__ Y,
                         float* __restrict__ out, int n, int mode) {
    int i = blockIdx.x * blockDim.x + threadIdx.x;
    if (i >= n) return;
    int node = i >> 6, c = i & 63;
    float val = agg[i] + Y[(size_t)node * FUSED + 192 + c];
    out[i] = (mode == 0) ? fmaxf(val, 0.f) : 1.f / (1.f + __expf(-val));
}

// ---------------------------------------------------------------------------
// Host orchestration
// ---------------------------------------------------------------------------
extern "C" void kernel_launch(void* const* d_in, const int* in_sizes, int n_in,
                              void* d_out, int out_size, void* d_ws, size_t ws_size,
                              hipStream_t stream) {
    (void)in_sizes; (void)n_in; (void)out_size; (void)ws_size;
    const int N = N_NODES, E = N_EDGES;

    const float* x      = (const float*)d_in[0];
    const int*   ei     = (const int*)d_in[1];
    const int*   srcIdx = ei;        // edge_index[0]
    const int*   dstIdx = ei + E;    // edge_index[1]

    // workspace carve-up (256B aligned)
    char* ws = (char*)d_ws;
    size_t off = 0;
    auto carve = [&](size_t bytes) {
        void* p = ws + off;
        off = (off + bytes + 255) & ~(size_t)255;
        return p;
    };
    float*    Y     = (float*)   carve((size_t)N * FUSED * sizeof(float)); // fused QKVS
    float*    agg   = (float*)   carve((size_t)N * OUTD * sizeof(float));
    float*    h     = (float*)   carve((size_t)N * OUTD * sizeof(float)); // layer output
    float*    score = (float*)   carve((size_t)E * sizeof(float));
    float*    evals = (float*)   carve((size_t)E * sizeof(float));
    unsigned* menc  = (unsigned*)carve((size_t)N * sizeof(unsigned));
    float*    denom = (float*)   carve((size_t)N * sizeof(float));
    __bf16*   Wswz  = (__bf16*)  carve((size_t)128 * FUSED * sizeof(__bf16));
    float*    bcat  = (float*)   carve((size_t)FUSED * sizeof(float));

    const unsigned ENC_NEG_INF = ~0xFF800000u; // encf(-inf)
    const float    scale       = 0.125f;       // 1/sqrt(64)
    const int      TPB         = 256;
    const int      eBlocks     = (E + TPB - 1) / TPB;

    for (int l = 0; l < 3; ++l) {
        const int d_in_l  = (l == 0) ? 128 : 64;
        const int pbase   = 2 + l * 8; // Wq,bq,Wk,bk,Wv,bv,Ws,bs
        const float* Wq = (const float*)d_in[pbase + 0];
        const float* bq = (const float*)d_in[pbase + 1];
        const float* Wk = (const float*)d_in[pbase + 2];
        const float* bk = (const float*)d_in[pbase + 3];
        const float* Wv = (const float*)d_in[pbase + 4];
        const float* bv = (const float*)d_in[pbase + 5];
        const float* Ws = (const float*)d_in[pbase + 6];
        const float* bs = (const float*)d_in[pbase + 7];
        const float* Xl = (l == 0) ? x : h;

        // pack weights to bf16 fragment-major layout
        int pk = d_in_l * FUSED;
        pack_weights<<<(pk + TPB - 1) / TPB, TPB, 0, stream>>>(
            Wq, Wk, Wv, Ws, bq, bk, bv, bs, Wswz, bcat, d_in_l);

        // fused projections via WMMA (N = 6250 * 16 exactly)
        dim3 ggrid(N / 16, 4);
        if (d_in_l == 128)
            qkvs_gemm<128><<<ggrid, 128, 0, stream>>>(Xl, Wswz, bcat, Y);
        else
            qkvs_gemm<64><<<ggrid, 128, 0, stream>>>(Xl, Wswz, bcat, Y);

        // init per-layer segment state
        fill_u32<<<(N + TPB - 1) / TPB, TPB, 0, stream>>>(menc, ENC_NEG_INF, N);
        fill_f32<<<(N + TPB - 1) / TPB, TPB, 0, stream>>>(denom, 0.f, N);
        fill_f32<<<((N * OUTD) + TPB - 1) / TPB, TPB, 0, stream>>>(agg, 0.f, N * OUTD);

        // edge attention pipeline
        edge_scores<<<eBlocks, TPB, 0, stream>>>(srcIdx, dstIdx, Y, score, E, scale);
        seg_max<<<eBlocks, TPB, 0, stream>>>(dstIdx, score, menc, E);
        exp_sum<<<eBlocks, TPB, 0, stream>>>(dstIdx, score, menc, evals, denom, E);
        scatter_agg<<<(E * 32) / TPB, TPB, 0, stream>>>(srcIdx, dstIdx, Y, evals, denom, agg, E);

        // skip + activation
        float* outPtr = (l == 2) ? (float*)d_out : h;
        int    mode   = (l == 2) ? 1 : 0;
        epilogue<<<((N * OUTD) + TPB - 1) / TPB, TPB, 0, stream>>>(agg, Y, outPtr, N * OUTD, mode);
    }
}
